// MultiHeadedAttention_46686294508012
// MI455X (gfx1250) — matrix operand order repaired and checked by reading, not compiled, not run on hardware
//
#include <hip/hip_runtime.h>
#include <hip/hip_bf16.h>
#include <math.h>
#include <stdint.h>

// ---------------------------------------------------------------------------
// MultiHeadedAttention (RoPE + clipped relative position bias, causal) for
// MI455X / gfx1250.  All matmuls use v_wmma_f32_16x16x32_f16 (f32 accum).
// Round 3: + async global->LDS staging (global_load_async_to_lds_b128 /
// s_wait_asynccnt) for the rel_emb table in the flash kernel.
//
// Shapes: B=2, L=1024, MODEL=1024, H=16, DH=64.  N = B*L = 2048.
// ---------------------------------------------------------------------------

typedef _Float16 h16 __attribute__((ext_vector_type(16)));
typedef _Float16 h8  __attribute__((ext_vector_type(8)));
typedef float    f8  __attribute__((ext_vector_type(8)));

// ---- WMMA fragment helpers (layouts per CDNA5 ISA 7.12.2) ------------------
// A (16x32 f16): lane l<16 -> row M=l, K elems {0..7,16..23}; l>=16 -> row l-16,
// K elems {8..15,24..31}.
__device__ __forceinline__ h16 load_a_frag(const _Float16* base, int stride) {
  const int l = threadIdx.x & 31;
  const int m = l & 15, g = l >> 4;
  const _Float16* p = base + m * stride + g * 8;
  h8 lo = *(const h8*)(p);
  h8 hi = *(const h8*)(p + 16);
  return __builtin_shufflevector(lo, hi, 0,1,2,3,4,5,6,7,8,9,10,11,12,13,14,15);
}

// B (32x16 f16): lane l -> col n=l&15; lanes 0-15 hold K={0..7,16..23}, lanes 16-31
// hold K={8..15,24..31}.  baseT points at (n0,k0) of B^T (row n = col of B).
__device__ __forceinline__ h16 load_b_frag(const _Float16* baseT, int stride) {
  const int l = threadIdx.x & 31;
  const int n = l & 15, g = l >> 4;
  const _Float16* p = baseT + n * stride + g * 8;
  h8 lo = *(const h8*)(p);
  h8 hi = *(const h8*)(p + 16);
  return __builtin_shufflevector(lo, hi, 0,1,2,3,4,5,6,7,8,9,10,11,12,13,14,15);
}

__device__ __forceinline__ f8 wmma_f16(h16 a, h16 b, f8 c) {
  return __builtin_amdgcn_wmma_f32_16x16x32_f16(false, a, false, b, (short)0, c,
                                                false, false);
}

// ---- f32 -> f16 conversion -------------------------------------------------
__global__ void mha_cvt_f16(const float* __restrict__ src,
                            _Float16* __restrict__ dst, int n) {
  int i = blockIdx.x * 256 + threadIdx.x;
  if (i < n) dst[i] = (_Float16)src[i];
}

// ---- Generic GEMM: C[m,n] = sum_k A[m,k] * BT[n,k];  M=2048, N=1024, K=1024.
// Block: 256 thr (8 waves) -> tile 128(M) x 64(N); wave w owns rows w*16..+15.
// Two-stage register double buffering: WMMAs consume fragments loaded one
// half-iteration earlier, so waits are partial (latency hidden by compute).
__global__ __launch_bounds__(256) void mha_gemm_nt(
    const _Float16* __restrict__ A, const _Float16* __restrict__ BT,
    float* __restrict__ C) {
  const int wave = threadIdx.x >> 5;
  const int row0 = blockIdx.x * 128 + wave * 16;
  const int col0 = blockIdx.y * 64;
  const _Float16* Ab = A + row0 * 1024;
  const _Float16* Bb = BT + col0 * 1024;

  f8 acc[4] = {{}, {}, {}, {}};
  h16 a0 = load_a_frag(Ab, 1024);
  h16 b0[4];
#pragma unroll
  for (int ns = 0; ns < 4; ++ns) b0[ns] = load_b_frag(Bb + ns * 16384, 1024);

  for (int k0 = 0; k0 < 1024; k0 += 64) {
    // prefetch stage 1 (k0+32, always in range)
    h16 a1 = load_a_frag(Ab + k0 + 32, 1024);
    h16 b1[4];
#pragma unroll
    for (int ns = 0; ns < 4; ++ns)
      b1[ns] = load_b_frag(Bb + ns * 16384 + k0 + 32, 1024);
    // compute stage 0
#pragma unroll
    for (int ns = 0; ns < 4; ++ns) acc[ns] = wmma_f16(a0, b0[ns], acc[ns]);
    // prefetch next stage 0 (k0+64)
    if (k0 + 64 < 1024) {
      a0 = load_a_frag(Ab + k0 + 64, 1024);
#pragma unroll
      for (int ns = 0; ns < 4; ++ns)
        b0[ns] = load_b_frag(Bb + ns * 16384 + k0 + 64, 1024);
    }
    // compute stage 1
#pragma unroll
    for (int ns = 0; ns < 4; ++ns) acc[ns] = wmma_f16(a1, b1[ns], acc[ns]);
  }

  const int l = threadIdx.x & 31, n = l & 15, g = l >> 4;
#pragma unroll
  for (int ns = 0; ns < 4; ++ns)
#pragma unroll
    for (int j = 0; j < 8; ++j)
      C[(row0 + 8 * g + j) * 1024 + col0 + ns * 16 + n] = acc[ns][j];
}

// ---- RoPE (interleaved) + 1/8 scale on Q; convert to f16; V -> [b,h,d,L] ---
__global__ void mha_rope_prep(const float* __restrict__ qf,
                              const float* __restrict__ kf,
                              const float* __restrict__ vf,
                              _Float16* __restrict__ qh,
                              _Float16* __restrict__ kh,
                              _Float16* __restrict__ vT) {
  int idx = blockIdx.x * 256 + threadIdx.x;        // one thread per (row, pair)
  if (idx >= 2048 * 512) return;
  int n   = idx >> 9;                              // row 0..2047 (b*L + l)
  int c0  = (idx & 511) * 2;                       // even column
  int pos = n & 1023;
  int i   = (c0 & 63) >> 1;                        // pair index within head
  float inv_freq = powf(10000.0f, -(float)i / 32.0f);
  float ang = (float)pos * inv_freq;
  float c = cosf(ang), s = sinf(ang);
  const float* q = qf + n * 1024 + c0;
  const float* k = kf + n * 1024 + c0;
  float qe = q[0], qo = q[1], ke = k[0], ko = k[1];
  const float sc = 0.125f;                         // 1/sqrt(64)
  qh[n * 1024 + c0]     = (_Float16)((qe * c - qo * s) * sc);
  qh[n * 1024 + c0 + 1] = (_Float16)((qe * s + qo * c) * sc);
  kh[n * 1024 + c0]     = (_Float16)(ke * c - ko * s);
  kh[n * 1024 + c0 + 1] = (_Float16)(ke * s + ko * c);
  int b = n >> 10, lpos = n & 1023;
  int h0 = c0 >> 6, d0 = c0 & 63;
  vT[((b * 16 + h0) * 64 + d0) * 1024 + lpos]     = (_Float16)vf[n * 1024 + c0];
  vT[((b * 16 + h0) * 64 + d0 + 1) * 1024 + lpos] = (_Float16)vf[n * 1024 + c0 + 1];
}

// ---- Flash attention with clipped relative bias ----------------------------
// Grid: 256 blocks = B(2) * H(16) * (L/128).  Block: 256 thr = 8 waves,
// wave w owns 16 query rows.  K-tile = 64.  V fragments and the next tile's
// K fragments are prefetched before the softmax VALU work so load latency
// overlaps with exp/shuffle/LDS band maintenance.  rel_emb is staged into
// LDS with gfx1250 async global->LDS DMA (ASYNCcnt-tracked).
__global__ __launch_bounds__(256) void mha_flash(
    const _Float16* __restrict__ qh, const _Float16* __restrict__ kh,
    const _Float16* __restrict__ vT, const float* __restrict__ rel,
    _Float16* __restrict__ ctx) {
  __shared__ __attribute__((aligned(16))) float    rel_s[33][64];
  __shared__ __attribute__((aligned(16))) float    qrel[8][16][33];
  __shared__ __attribute__((aligned(16))) float    band[8][16][32];
  __shared__ __attribute__((aligned(16))) _Float16 pl[8][16][64];
  __shared__ float scalA[8][16];
  __shared__ float scalB[8][16];

  const int tid = threadIdx.x;
  const int wave = tid >> 5, l = tid & 31;
  const int lane16 = l & 15, g = l >> 4;

  const int bid = blockIdx.x;
  const int qblk = bid & 7, h = (bid >> 3) & 15, b = bid >> 7;
  const int qw0 = qblk * 128 + wave * 16;          // first query row of wave
  const int rowBase = b * 1024 + qw0;              // row into [N,1024] tensors
  const int colOff = h * 64;

  // Stage rel_emb rows 0..32 (33*64 f32 = 528 x 16B) into LDS via async DMA.
  // LDS operand = low 32 bits of the shared-aperture flat address.
  {
    const uint32_t ldsBase = (uint32_t)(uintptr_t)&rel_s[0][0];
    for (int t = tid; t < 528; t += 256) {
      uint64_t ga = (uint64_t)(uintptr_t)rel + (uint64_t)t * 16u;
      uint32_t la = ldsBase + (uint32_t)t * 16u;
      asm volatile("global_load_async_to_lds_b128 %0, %1, off"
                   :: "v"(la), "v"(ga) : "memory");
    }
    asm volatile("s_wait_asynccnt 0x0" ::: "memory");
  }
  __syncthreads();

  // qrel[row][r] = q_scaled_roped[row] . rel_emb[r],  r = 0..32
  {
    const _Float16* qrow = qh + (rowBase + lane16) * 1024 + colOff;
    for (int r = g; r < 33; r += 2) {
      float s = 0.f;
      for (int d = 0; d < 64; ++d) s += (float)qrow[d] * rel_s[r][d];
      qrel[wave][lane16][r] = s;
    }
    for (int t = 0; t < 16; ++t) band[wave][lane16][g * 16 + t] = 0.f;
  }

  float m[8], lsum[8];
  f8 o[4] = {{}, {}, {}, {}};
#pragma unroll
  for (int j = 0; j < 8; ++j) { m[j] = -3.0e38f; lsum[j] = 0.f; }

  const h16 aq0 = load_a_frag(qh + rowBase * 1024 + colOff, 1024);
  const h16 aq1 = load_a_frag(qh + rowBase * 1024 + colOff + 32, 1024);

  h16 bk[8];
  auto load_ktile = [&](int kk) {
#pragma unroll
    for (int ns = 0; ns < 4; ++ns) {
      const _Float16* kb = kh + (b * 1024 + kk + ns * 16) * 1024 + colOff;
      bk[2 * ns]     = load_b_frag(kb, 1024);
      bk[2 * ns + 1] = load_b_frag(kb + 32, 1024);
    }
  };
  load_ktile(0);

  const int kend = qw0 + 16;                       // exclusive causal bound
  for (int k0 = 0; k0 < kend; k0 += 64) {
    // S = Q @ K^T  (16 rows x 64 cols per wave)
    f8 S[4];
#pragma unroll
    for (int ns = 0; ns < 4; ++ns) {
      f8 z = {};
      f8 t = wmma_f16(aq0, bk[2 * ns], z);
      S[ns] = wmma_f16(aq1, bk[2 * ns + 1], t);
    }
    // prefetch V fragments for this tile (consumed after softmax)
    h16 bv[8];
#pragma unroll
    for (int ds = 0; ds < 4; ++ds) {
      const _Float16* vb = vT + ((b * 16 + h) * 64 + ds * 16) * 1024 + k0;
      bv[2 * ds]     = load_b_frag(vb, 1024);
      bv[2 * ds + 1] = load_b_frag(vb + 32, 1024);
    }
    // prefetch next tile's K fragments
    if (k0 + 64 < kend) load_ktile(k0 + 64);

    // bias + causal mask + online softmax
    float newm[8];
#pragma unroll
    for (int j = 0; j < 8; ++j) {
      const int M = j + 8 * g, q = qw0 + M;
      float mx = -3.0e38f;
#pragma unroll
      for (int ns = 0; ns < 4; ++ns) {
        int k = k0 + ns * 16 + lane16;
        int idx = k - q + 32;
        idx = idx < 0 ? 0 : (idx > 32 ? 32 : idx);
        float s = S[ns][j] + qrel[wave][M][idx];
        if (k > q) s = -1.0e18f;
        S[ns][j] = s;
        mx = fmaxf(mx, s);
      }
      for (int off = 1; off < 16; off <<= 1) mx = fmaxf(mx, __shfl_xor(mx, off));
      newm[j] = fmaxf(m[j], mx);
    }
#pragma unroll
    for (int j = 0; j < 8; ++j) {
      float alpha = __expf(m[j] - newm[j]);
      m[j] = newm[j];
      float rs = 0.f;
#pragma unroll
      for (int ns = 0; ns < 4; ++ns) {
        float p = __expf(S[ns][j] - newm[j]);
        S[ns][j] = p;
        rs += p;
      }
      for (int off = 1; off < 16; off <<= 1) rs += __shfl_xor(rs, off);
      lsum[j] = lsum[j] * alpha + rs;
#pragma unroll
      for (int ns = 0; ns < 4; ++ns) o[ns][j] *= alpha;
      if (l == 0)  scalA[wave][j] = alpha;
      if (l == 16) scalA[wave][8 + j] = alpha;
    }
    asm volatile("s_wait_dscnt 0x0" ::: "memory");
    { // rescale the diagonal band by alpha(row)
      float a = scalA[wave][lane16];
      for (int t = 0; t < 16; ++t) band[wave][lane16][g * 16 + t] *= a;
    }
    // stage P to LDS (A-layout refill) and record band entries
#pragma unroll
    for (int j = 0; j < 8; ++j) {
      const int M = j + 8 * g, q = qw0 + M;
#pragma unroll
      for (int ns = 0; ns < 4; ++ns) {
        int k = k0 + ns * 16 + lane16;
        float p = S[ns][j];
        pl[wave][M][k - k0] = (_Float16)p;
        int r1 = k - q + 31;                       // bucket r-1 (r=1..32)
        if (r1 >= 0 && r1 < 32) band[wave][M][r1] = p;
      }
    }
    asm volatile("s_wait_dscnt 0x0" ::: "memory");
    // O += P @ V  (V fragments already in registers)
    const h16 ap0 = load_a_frag(&pl[wave][0][0], 64);
    const h16 ap1 = load_a_frag(&pl[wave][0][32], 64);
#pragma unroll
    for (int ds = 0; ds < 4; ++ds) {
      o[ds] = wmma_f16(ap0, bv[2 * ds], o[ds]);
      o[ds] = wmma_f16(ap1, bv[2 * ds + 1], o[ds]);
    }
  }

  // finalize: normalize O and band, add rel-context term
  float invl[8];
#pragma unroll
  for (int j = 0; j < 8; ++j) {
    invl[j] = 1.0f / lsum[j];
#pragma unroll
    for (int ns = 0; ns < 4; ++ns) o[ns][j] *= invl[j];
    if (l == 0)  scalA[wave][j] = invl[j];
    if (l == 16) scalA[wave][8 + j] = invl[j];
  }
  asm volatile("s_wait_dscnt 0x0" ::: "memory");
  {
    float a = scalA[wave][lane16];
    float part = 0.f;
    for (int t = 0; t < 16; ++t) {
      float v = band[wave][lane16][g * 16 + t] * a;
      band[wave][lane16][g * 16 + t] = v;
      part += v;
    }
    part += __shfl_xor(part, 16);
    if (g == 0) scalB[wave][lane16] = part;        // sum of band mass per row
  }
  asm volatile("s_wait_dscnt 0x0" ::: "memory");
  // O += band_norm(16x32) @ rel_emb[1..32](32x64)   (one WMMA per 16 dims)
  h16 ab = {};
#pragma unroll
  for (int e = 0; e < 8; ++e) {
    ab[e]     = (_Float16)band[wave][lane16][g * 8 + e];
    ab[e + 8] = (_Float16)band[wave][lane16][16 + g * 8 + e];
  }
#pragma unroll
  for (int ds = 0; ds < 4; ++ds) {
    h16 bb = {};
#pragma unroll
    for (int e = 0; e < 16; ++e)
      bb[e] = (_Float16)rel_s[1 + (e < 8 ? g * 8 + e : 16 + g * 8 + (e - 8))][ds * 16 + lane16];
    o[ds] = wmma_f16(ab, bb, o[ds]);
  }
  // bucket r=0 gets the remaining mass: (1 - sum(band)) * rel_emb[0]
#pragma unroll
  for (int j = 0; j < 8; ++j) {
    const int M = j + 8 * g;
    float a0 = 1.0f - scalB[wave][M];
#pragma unroll
    for (int ns = 0; ns < 4; ++ns) {
      float val = o[ns][j] + a0 * rel_s[0][ns * 16 + lane16];
      ctx[(rowBase + M) * 1024 + colOff + ns * 16 + lane16] = (_Float16)val;
    }
  }
}

// ---------------------------------------------------------------------------
extern "C" void kernel_launch(void* const* d_in, const int* in_sizes, int n_in,
                              void* d_out, int out_size, void* d_ws,
                              size_t ws_size, hipStream_t stream) {
  const float* x   = (const float*)d_in[0];
  const float* wq  = (const float*)d_in[1];
  const float* wk  = (const float*)d_in[2];
  const float* wv  = (const float*)d_in[3];
  const float* wo  = (const float*)d_in[4];
  const float* rel = (const float*)d_in[5];

  char* ws = (char*)d_ws;                          // ~52 MB used
  _Float16* xh   = (_Float16*)(ws);                // [2048,1024] f16  (4 MB)
  _Float16* wqh  = (_Float16*)(ws + (4u  << 20));  // 2 MB each
  _Float16* wkh  = (_Float16*)(ws + (6u  << 20));
  _Float16* wvh  = (_Float16*)(ws + (8u  << 20));
  _Float16* woh  = (_Float16*)(ws + (10u << 20));
  float*    qf   = (float*)   (ws + (12u << 20));  // [2048,1024] f32 (8 MB each)
  float*    kf   = (float*)   (ws + (20u << 20));
  float*    vf   = (float*)   (ws + (28u << 20));
  _Float16* qhh  = (_Float16*)(ws + (36u << 20));  // roped+scaled q, f16
  _Float16* khh  = (_Float16*)(ws + (40u << 20));  // roped k, f16
  _Float16* vT   = (_Float16*)(ws + (44u << 20));  // [b,h,64,L] f16
  _Float16* ctxh = (_Float16*)(ws + (48u << 20));  // [2048,1024] f16

  const int NX = 2048 * 1024, NW = 1024 * 1024;
  mha_cvt_f16<<<(NX + 255) / 256, 256, 0, stream>>>(x,  xh,  NX);
  mha_cvt_f16<<<(NW + 255) / 256, 256, 0, stream>>>(wq, wqh, NW);
  mha_cvt_f16<<<(NW + 255) / 256, 256, 0, stream>>>(wk, wkh, NW);
  mha_cvt_f16<<<(NW + 255) / 256, 256, 0, stream>>>(wv, wvh, NW);
  mha_cvt_f16<<<(NW + 255) / 256, 256, 0, stream>>>(wo, woh, NW);

  dim3 gg(16, 16);                                 // 2048/128 x 1024/64
  mha_gemm_nt<<<gg, 256, 0, stream>>>(xh, wqh, qf);
  mha_gemm_nt<<<gg, 256, 0, stream>>>(xh, wkh, kf);
  mha_gemm_nt<<<gg, 256, 0, stream>>>(xh, wvh, vf);

  mha_rope_prep<<<(2048 * 512 + 255) / 256, 256, 0, stream>>>(qf, kf, vf,
                                                              qhh, khh, vT);

  mha_flash<<<256, 256, 0, stream>>>(qhh, khh, vT, rel, ctxh);

  mha_gemm_nt<<<gg, 256, 0, stream>>>(ctxh, woh, (float*)d_out);
}
